// CasualAttention_30374008717797
// MI455X (gfx1250) — compile-verified, matrix operand order checked by reading, not run on hardware
//
#include <hip/hip_runtime.h>
#include <math.h>

// Problem constants (from reference setup_inputs)
constexpr int B = 4;
constexpr int S = 2048;
constexpr int E = 1024;
constexpr int D = 1024;

typedef __attribute__((ext_vector_type(16))) _Float16 v16h;
typedef __attribute__((ext_vector_type(8)))  _Float16 v8h;
typedef __attribute__((ext_vector_type(8)))  float    v8f;

// ---------------------------------------------------------------------------
// WMMA fragment loader (wave32, v_wmma_f32_16x16x32_f16)
// A-matrix 16x32 f16 per-lane layout (ISA 7.12.2):
//   lane<16 : row M=lane,    halves[0..7]=K 0..7,  halves[8..15]=K 16..23
//   lane>=16: row M=lane-16, halves[0..7]=K 8..15, halves[8..15]=K 24..31
// B-matrix (32x16) mirrors this with lane <-> column N, so row-major [N][K]
// (nn.Linear weight layout, k rows, vT rows) loads with the SAME pattern.
// ---------------------------------------------------------------------------
static __device__ inline v16h load_frag_f16(const _Float16* __restrict__ base,
                                            int pitch, int row0, int k0, int lane) {
    const int m   = lane & 15;
    const int klo = (lane >> 4) << 3;  // 0 or 8
    const _Float16* p = base + (size_t)(row0 + m) * pitch + (k0 + klo);
    v8h lo = *(const v8h*)(p);
    v8h hi = *(const v8h*)(p + 16);
    return __builtin_shufflevector(lo, hi, 0, 1, 2, 3, 4, 5, 6, 7,
                                           8, 9, 10, 11, 12, 13, 14, 15);
}

static __device__ inline v8f wmma_f16(v16h a, v16h b, v8f c) {
    return __builtin_amdgcn_wmma_f32_16x16x32_f16(
        /*neg_a=*/false, a, /*neg_b=*/false, b,
        /*c_mod=*/(short)0, c, /*reuse_a=*/false, /*reuse_b=*/false);
}

static __device__ inline v8f zero8f() {
    v8f z = {0.f, 0.f, 0.f, 0.f, 0.f, 0.f, 0.f, 0.f};
    return z;
}

// C/D f32 16x16 layout: VGPR r holds M = r + 8*(lane>=16), column N = lane&15
static __device__ inline int cd_row(int r, int lane) { return r + ((lane >> 4) << 3); }

// ---------------------------------------------------------------------------
// Pipelined 16x128 GEMM fragment set: 1 A frag reused by 8 WMMAs.
// Register budget: acc 64 + 2 sets * 72 = ~210 VGPRs -> fits under 256,
// so BOTH pipeline halves can keep their loads in flight (staggered waits).
// ---------------------------------------------------------------------------
struct FragAB8 { v16h a; v16h b[8]; };

static __device__ inline void ab8_load(FragAB8& f,
    const _Float16* __restrict__ abase, int apitch, int arow0,
    const _Float16* __restrict__ bbase, int bpitch, int brow0,
    int k0, int lane) {
    f.a = load_frag_f16(abase, apitch, arow0, k0, lane);
#pragma unroll
    for (int u = 0; u < 8; ++u)
        f.b[u] = load_frag_f16(bbase, bpitch, brow0 + 16 * u, k0, lane);
}

static __device__ inline void ab8_mma(const FragAB8& f, v8f acc[8]) {
#pragma unroll
    for (int u = 0; u < 8; ++u) acc[u] = wmma_f16(f.a, f.b[u], acc[u]);
}

// ---------------------------------------------------------------------------
// Kernel 0: vectorized f32 -> f16 conversion (8 elems/thread/iter)
// ---------------------------------------------------------------------------
__global__ void cvt_f32_to_f16_kernel(const float* __restrict__ in,
                                      _Float16* __restrict__ out, int n8) {
    int i = blockIdx.x * blockDim.x + threadIdx.x;
    int stride = gridDim.x * blockDim.x;
    for (; i < n8; i += stride) {
        v8f v = ((const v8f*)in)[i];
        ((v8h*)out)[i] = __builtin_convertvector(v, v8h);
    }
}

// ---------------------------------------------------------------------------
// Kernel 1: projection GEMM  y[m][d] = sum_e x[m][e] * w[d][e].
// One wave per 16(M) x 128(N) tile, 2-stage software pipeline over K.
// TR=false: y row-major f16 [B*S][D] (q, k).
// TR=true : y transposed f16 [B][D][S] (v), so attn@v B-frags are contiguous.
// ---------------------------------------------------------------------------
template <bool TR>
__global__ __launch_bounds__(256) void proj_gemm_kernel(
    const _Float16* __restrict__ xh,  // [B*S][E]
    const _Float16* __restrict__ wh,  // [D][E]
    _Float16* __restrict__ yh) {
    const int lane = threadIdx.x & 31;
    const int wid  = blockIdx.x * 8 + (threadIdx.x >> 5);
    const int row0 = (wid >> 3) * 16;   // 512 M-tiles over B*S
    const int d0   = (wid & 7) * 128;   // 8 N128-tiles over D

    v8f acc[8];
#pragma unroll
    for (int u = 0; u < 8; ++u) acc[u] = zero8f();

    FragAB8 f0, f1;
    ab8_load(f0, xh, E, row0, wh, E, d0, 0, lane);
    for (int k0 = 0; k0 < E; k0 += 64) {
        if (k0 + 32 < E) ab8_load(f1, xh, E, row0, wh, E, d0, k0 + 32, lane);
        ab8_mma(f0, acc);
        if (k0 + 64 < E) ab8_load(f0, xh, E, row0, wh, E, d0, k0 + 64, lane);
        if (k0 + 32 < E) ab8_mma(f1, acc);
    }

    const int n    = lane & 15;
    const int bidx = row0 / S;  // 16-row tiles never straddle a batch
    const int sloc = row0 % S;
#pragma unroll
    for (int u = 0; u < 8; ++u) {
#pragma unroll
        for (int r = 0; r < 8; ++r) {
            const int m = cd_row(r, lane);
            const int d = d0 + 16 * u + n;
            if (TR)
                yh[((size_t)bidx * D + d) * S + (sloc + m)] = (_Float16)acc[u][r];
            else
                yh[(size_t)(row0 + m) * D + d] = (_Float16)acc[u][r];
        }
    }
}

// ---------------------------------------------------------------------------
// Kernel 2: causal scores, 16(s) x 128(t) tile per wave. Rectangular launch
// over (b, i, jj) with uniform per-wave early-exit for tiles fully above the
// diagonal. scores = (q . k) / sqrt(D); masked to -inf where t > s.
// ---------------------------------------------------------------------------
__global__ __launch_bounds__(256) void scores_gemm_kernel(
    const _Float16* __restrict__ qh, const _Float16* __restrict__ kh,
    float* __restrict__ scores) {
    const int lane = threadIdx.x & 31;
    const int wid  = blockIdx.x * 8 + (threadIdx.x >> 5);
    const int b    = wid / (128 * 16);
    const int rem  = wid % (128 * 16);
    const int i    = rem >> 4;          // query 16-tile
    const int jj   = rem & 15;          // key 128-tile
    const int s0   = i * 16;
    const int t0   = jj * 128;
    if (t0 > s0 + 15) return;           // tile entirely above diagonal

    const _Float16* qb = qh + (size_t)b * S * D;
    const _Float16* kb = kh + (size_t)b * S * D;

    v8f acc[8];
#pragma unroll
    for (int u = 0; u < 8; ++u) acc[u] = zero8f();

    FragAB8 f0, f1;
    ab8_load(f0, qb, D, s0, kb, D, t0, 0, lane);
    for (int k0 = 0; k0 < D; k0 += 64) {
        if (k0 + 32 < D) ab8_load(f1, qb, D, s0, kb, D, t0, k0 + 32, lane);
        ab8_mma(f0, acc);
        if (k0 + 64 < D) ab8_load(f0, qb, D, s0, kb, D, t0, k0 + 64, lane);
        if (k0 + 32 < D) ab8_mma(f1, acc);
    }

    const float inv_sqrt_d = 0.03125f;  // 1/sqrt(1024)
    const int n = lane & 15;
#pragma unroll
    for (int u = 0; u < 8; ++u) {
#pragma unroll
        for (int r = 0; r < 8; ++r) {
            const int s = s0 + cd_row(r, lane);
            const int t = t0 + 16 * u + n;
            float v = acc[u][r] * inv_sqrt_d;
            if (t > s) v = -INFINITY;
            scores[((size_t)b * S + s) * S + t] = v;
        }
    }
}

// ---------------------------------------------------------------------------
// Kernel 3: row softmax over t in [0, s]; writes f16 probabilities, zero-
// padded up to the next multiple of 32 so the output GEMM reads whole K=32
// chunks of defined data.
// ---------------------------------------------------------------------------
__global__ __launch_bounds__(256) void softmax_rows_kernel(
    const float* __restrict__ scores, _Float16* __restrict__ probs) {
    __shared__ float red[256];
    const int row = blockIdx.x;          // b*S + s
    const int s   = row & (S - 1);
    const int tid = threadIdx.x;
    const float* srow = scores + (size_t)row * S;
    _Float16*    prow = probs  + (size_t)row * S;

    float lm = -INFINITY;
    for (int t = tid; t <= s; t += 256) lm = fmaxf(lm, srow[t]);
    red[tid] = lm; __syncthreads();
    for (int off = 128; off > 0; off >>= 1) {
        if (tid < off) red[tid] = fmaxf(red[tid], red[tid + off]);
        __syncthreads();
    }
    const float m = red[0]; __syncthreads();

    float ls = 0.f;
    for (int t = tid; t <= s; t += 256) ls += __expf(srow[t] - m);
    red[tid] = ls; __syncthreads();
    for (int off = 128; off > 0; off >>= 1) {
        if (tid < off) red[tid] += red[tid + off];
        __syncthreads();
    }
    const float inv = 1.0f / red[0];

    const int zend = min(S, ((s + 32) >> 5) << 5);  // round-up-32(s+1)
    for (int t = tid; t < zend; t += 256) {
        float pv = (t <= s) ? __expf(srow[t] - m) * inv : 0.f;
        prow[t] = (_Float16)pv;
    }
}

// ---------------------------------------------------------------------------
// Kernel 4: out = probs @ v. One wave per 16(M) x 128(N) tile, K bound per
// query tile i is ceil(16*(i+1)/32) chunks (causal). B fragments come from
// vT [B][D][S] so the loader pattern is contiguous. Pipelined.
// ---------------------------------------------------------------------------
__global__ __launch_bounds__(256) void out_gemm_kernel(
    const _Float16* __restrict__ probs,  // [B][S][S]
    const _Float16* __restrict__ vTh,    // [B][D][S]
    float* __restrict__ out) {           // [B][S][D]
    const int lane = threadIdx.x & 31;
    const int wid  = blockIdx.x * 8 + (threadIdx.x >> 5);
    const int b    = wid / (128 * 8);
    const int rem  = wid % (128 * 8);
    const int i    = rem >> 3;            // query 16-tile
    const int s0   = i * 16;
    const int d0   = (rem & 7) * 128;
    const int kend = ((i + 2) >> 1) * 32; // ceil(16*(i+1)/32) K chunks

    const _Float16* pb = probs + (size_t)b * S * S;
    const _Float16* vb = vTh   + (size_t)b * D * S;

    v8f acc[8];
#pragma unroll
    for (int u = 0; u < 8; ++u) acc[u] = zero8f();

    FragAB8 f0, f1;
    ab8_load(f0, pb, S, s0, vb, S, d0, 0, lane);
    for (int k0 = 0; k0 < kend; k0 += 64) {
        if (k0 + 32 < kend) ab8_load(f1, pb, S, s0, vb, S, d0, k0 + 32, lane);
        ab8_mma(f0, acc);
        if (k0 + 64 < kend) ab8_load(f0, pb, S, s0, vb, S, d0, k0 + 64, lane);
        if (k0 + 32 < kend) ab8_mma(f1, acc);
    }

    const int n = lane & 15;
#pragma unroll
    for (int u = 0; u < 8; ++u) {
#pragma unroll
        for (int r = 0; r < 8; ++r) {
            const int m = cd_row(r, lane);
            out[((size_t)b * S + (s0 + m)) * D + d0 + 16 * u + n] = acc[u][r];
        }
    }
}

// ---------------------------------------------------------------------------
// Host launcher. Workspace budget (~174 MB):
//   x_h 16MB | wq/wk/wv_h 2MB each | q_h,k_h,vT_h 16MB each |
//   scores 67MB (f32) | probs 33.5MB (f16)
// ---------------------------------------------------------------------------
extern "C" void kernel_launch(void* const* d_in, const int* in_sizes, int n_in,
                              void* d_out, int out_size, void* d_ws, size_t ws_size,
                              hipStream_t stream) {
    const float* x  = (const float*)d_in[0];
    const float* wq = (const float*)d_in[1];
    const float* wk = (const float*)d_in[2];
    const float* wv = (const float*)d_in[3];
    float* out = (float*)d_out;

    size_t off = 0;
    auto alloc = [&](size_t bytes) -> void* {
        void* p = (char*)d_ws + off;
        off += (bytes + 255) & ~(size_t)255;
        return p;
    };
    _Float16* xh  = (_Float16*)alloc((size_t)B * S * E * 2);
    _Float16* wqh = (_Float16*)alloc((size_t)D * E * 2);
    _Float16* wkh = (_Float16*)alloc((size_t)D * E * 2);
    _Float16* wvh = (_Float16*)alloc((size_t)D * E * 2);
    _Float16* qh  = (_Float16*)alloc((size_t)B * S * D * 2);
    _Float16* kh  = (_Float16*)alloc((size_t)B * S * D * 2);
    _Float16* vTh = (_Float16*)alloc((size_t)B * D * S * 2);
    float*    sc  = (float*)   alloc((size_t)B * S * S * 4);
    _Float16* ph  = (_Float16*)alloc((size_t)B * S * S * 2);
    (void)ws_size; (void)in_sizes; (void)n_in; (void)out_size;

    // 0) conversions (vectorized x8)
    cvt_f32_to_f16_kernel<<<1024, 256, 0, stream>>>(x, xh, B * S * E / 8);
    cvt_f32_to_f16_kernel<<<512, 256, 0, stream>>>(wq, wqh, D * E / 8);
    cvt_f32_to_f16_kernel<<<512, 256, 0, stream>>>(wk, wkh, D * E / 8);
    cvt_f32_to_f16_kernel<<<512, 256, 0, stream>>>(wv, wvh, D * E / 8);

    // 1) projections: (B*S/16) * (D/128) = 4096 waves -> 512 blocks each
    proj_gemm_kernel<false><<<512, 256, 0, stream>>>(xh, wqh, qh);
    proj_gemm_kernel<false><<<512, 256, 0, stream>>>(xh, wkh, kh);
    proj_gemm_kernel<true><<<512, 256, 0, stream>>>(xh, wvh, vTh);

    // 2) causal scores: B * 128 * 16 = 8192 waves -> 1024 blocks (early exit)
    scores_gemm_kernel<<<1024, 256, 0, stream>>>(qh, kh, sc);

    // 3) row softmax: one block per (b, s) row
    softmax_rows_kernel<<<B * S, 256, 0, stream>>>(sc, ph);

    // 4) attn @ v: B * 128 * 8 = 4096 waves -> 512 blocks
    out_gemm_kernel<<<512, 256, 0, stream>>>(ph, vTh, out);
}
